// tinyRSNN_49615462203515
// MI455X (gfx1250) — matrix operand order repaired and checked by reading, not compiled, not run on hardware
//
#include <hip/hip_runtime.h>
#include <math.h>

typedef __attribute__((ext_vector_type(16))) __bf16 v16bf;
typedef __attribute__((ext_vector_type(8)))  __bf16 v8bf;
typedef __attribute__((ext_vector_type(8)))  float  v8f;
typedef __attribute__((ext_vector_type(4)))  float  v4f;

#define NT   4096
#define NB   128
#define NIN  96
#define NH   64
#define NOUT 2

static __device__ inline v8f wmma_bf16(v16bf a, v16bf b, v8f c) {
  // D = A(16x32 bf16) * B(32x16 bf16) + C(16x16 f32)
  return __builtin_amdgcn_wmma_f32_16x16x32_bf16(
      /*neg_a=*/false, a, /*neg_b=*/false, b,
      /*c_mod=*/(short)0, c, /*reuse_a=*/false, /*reuse_b=*/false);
}

// B-fragment: per-lane 16 consecutive f32 -> bf16 (k = 32c + 16*half + e)
static __device__ inline v16bf bf16_from_f32x16(const float* __restrict__ p) {
  v16bf r;
#pragma unroll
  for (int i = 0; i < 16; i += 4) {
    v4f f = *(const v4f*)(p + i);
#pragma unroll
    for (int q = 0; q < 4; ++q) r[i + q] = (__bf16)f[q];
  }
  return r;
}

// A-fragment from LDS (bf16): elements 0..15 = k0..k0+15
static __device__ inline v16bf afrag_from_lds(const __bf16* rowbase, int k0) {
  v8bf lo = *(const v8bf*)(rowbase + k0);
  v8bf hi = *(const v8bf*)(rowbase + k0 + 8);
  v16bf r;
#pragma unroll
  for (int i = 0; i < 8; ++i) { r[i] = lo[i]; r[8 + i] = hi[i]; }
  return r;
}

// A-fragment from global f32 row (convert to bf16)
static __device__ inline v16bf afrag_from_global(const float* __restrict__ rowbase, int k0) {
  v16bf r;
#pragma unroll
  for (int g = 0; g < 2; ++g) {
    const float* p = rowbase + k0 + g * 8;
    v4f f0 = *(const v4f*)(p);
    v4f f1 = *(const v4f*)(p + 4);
#pragma unroll
    for (int q = 0; q < 4; ++q) {
      r[8 * g + q]     = (__bf16)f0[q];
      r[8 * g + 4 + q] = (__bf16)f1[q];
    }
  }
  return r;
}

// ---------------------------------------------------------------------------
// Parallel precompute: cur[t] = x[t-1] @ Wh^T for all t, whole-GPU WMMA GEMM.
// Output stored in D-fragment layout: ws[((t*8 + mi)*4 + ni)*32 + lane][0..7]
// so the scan kernel reloads each 16x16 tile as one contiguous v8f per lane.
// ---------------------------------------------------------------------------
__global__ __launch_bounds__(256)
void tinyrsnn_input_gemm(const float* __restrict__ x,
                         const float* __restrict__ Wh,
                         float* __restrict__ ws)
{
  const int t    = blockIdx.x;          // 0..NT-1
  const int mi   = threadIdx.x >> 5;    // 8 waves = 8 M-tiles
  const int lane = threadIdx.x & 31;
  const int n    = lane & 15;
  const int half = lane >> 4;

  float* dst = ws + ((((size_t)t * 8 + mi) * 4) * 32 + lane) * 8;

  if (t == 0) {                          // step 0 has no previous input
    v8f z = {};
#pragma unroll
    for (int ni = 0; ni < 4; ++ni) *(v8f*)(dst + (size_t)ni * 32 * 8) = z;
    return;
  }

  const float* xrow = x + (size_t)(t - 1) * NB * NIN + (size_t)(16 * mi + n) * NIN;
  v16bf xf[3];
#pragma unroll
  for (int c = 0; c < 3; ++c) xf[c] = afrag_from_global(xrow, 32 * c + half * 16);

#pragma unroll
  for (int ni = 0; ni < 4; ++ni) {
    const float* wr = Wh + (size_t)(16 * ni + n) * NIN;
    v16bf b0 = bf16_from_f32x16(wr + half * 16);
    v16bf b1 = bf16_from_f32x16(wr + 32 + half * 16);
    v16bf b2 = bf16_from_f32x16(wr + 64 + half * 16);
    v8f acc = {};
    acc = wmma_bf16(xf[0], b0, acc);
    acc = wmma_bf16(xf[1], b1, acc);
    acc = wmma_bf16(xf[2], b2, acc);
    *(v8f*)(dst + (size_t)ni * 32 * 8) = acc;
  }
}

// ---------------------------------------------------------------------------
// Persistent sequential scan: one workgroup, 16 wave32s, 2 N-tiles per wave.
// Post-barrier critical path is only: ds_load spikes -> 2 WMMA -> update.
// The input contribution for step t+1 is produced BEFORE the barrier (either
// loaded from ws fragments or computed with 3 pipelined WMMAs), seeded with
// syn*alpha so no extra vector add is needed on the critical path.
// ---------------------------------------------------------------------------
template <bool USE_WS>
__global__ __launch_bounds__(512)
void tinyrsnn_scan(const float* __restrict__ x,
                   const float* __restrict__ Wh,
                   const float* __restrict__ Vw,
                   const float* __restrict__ Wo,
                   const float* __restrict__ cur,   // ws fragments (USE_WS)
                   float* __restrict__ out,
                   float a_h, float b_h, float a_o, float b_o)
{
  __shared__ __align__(16) __bf16 spk[2][NB][NH];   // double-buffered spikes

  const int tid  = threadIdx.x;
  const int wave = tid >> 5;
  const int lane = tid & 31;
  const int n    = lane & 15;
  const int half = lane >> 4;
  const int mi   = wave >> 1;
  const int ni0  = (wave & 1) * 2;
  const bool do_out = (ni0 == 0);

  for (int i = tid; i < 2 * NB * NH; i += 512)
    ((__bf16*)spk)[i] = (__bf16)0.0f;

  // constant B fragments
  v16bf vf[2][2];
  v16bf whf[2][3];
#pragma unroll
  for (int j = 0; j < 2; ++j) {
    const float* vr = Vw + (size_t)(16 * (ni0 + j) + n) * NH;
#pragma unroll
    for (int c = 0; c < 2; ++c) vf[j][c] = bf16_from_f32x16(vr + 32 * c + half * 16);
    if (!USE_WS) {
      const float* wr = Wh + (size_t)(16 * (ni0 + j) + n) * NIN;
#pragma unroll
      for (int c = 0; c < 3; ++c) whf[j][c] = bf16_from_f32x16(wr + 32 * c + half * 16);
    }
  }
  v16bf wof[2];
  {
    const float* wr = Wo + (size_t)((n < NOUT) ? n : 0) * NH;
#pragma unroll
    for (int c = 0; c < 2; ++c) {
      v16bf f;
#pragma unroll
      for (int e = 0; e < 16; ++e) {
        float vv = (n < NOUT) ? wr[32 * c + half * 16 + e] : 0.0f;
        f[e] = (__bf16)vv;
      }
      wof[c] = f;
    }
  }

  v8f syn[2] = {}, mem[2] = {};
  v8f syno = {}, memo = {};
  v8f curN[2] = {};                 // alpha*syn + input-drive for NEXT step
  const float omb_h = 1.0f - b_h;
  const float omb_o = 1.0f - b_o;

  __syncthreads();

  for (int t = 0; t < NT; ++t) {
    const int rb = (t + 1) & 1;     // spikes from step t-1
    const int wb = t & 1;           // spikes produced this step

    // ---- post-barrier critical path -------------------------------------
    const __bf16* srow = &spk[rb][16 * mi + n][0];
    v16bf sfa0 = afrag_from_lds(srow, half * 16);
    v16bf sfa1 = afrag_from_lds(srow, 32 + half * 16);

#pragma unroll
    for (int j = 0; j < 2; ++j) {
      v8f acc = curN[j];                       // = alpha*syn + x@Wh^T (ready)
      acc = wmma_bf16(sfa0, vf[j][0], acc);
      acc = wmma_bf16(sfa1, vf[j][1], acc);
      syn[j] = acc;
      v8f mm = mem[j] * b_h + acc * omb_h;
#pragma unroll
      for (int r = 0; r < 8; ++r) {
        float u = mm[r];
        float s = (u > 1.0f) ? 1.0f : 0.0f;    // Heaviside(mem - 1)
        mm[r] = u - u * s;                     // reset-to-zero on spike
        spk[wb][16 * mi + half * 8 + r][16 * (ni0 + j) + n] = (__bf16)s;
      }
      mem[j] = mm;
    }

    if (do_out) {                              // readout LIF (prev spikes)
      v8f acc = syno * a_o;
      acc = wmma_bf16(sfa0, wof[0], acc);
      acc = wmma_bf16(sfa1, wof[1], acc);
      syno = acc;
      memo = memo * b_o + acc * omb_o;
      if (n < NOUT) {
        float* orow = out + (size_t)t * NB * NOUT;
#pragma unroll
        for (int r = 0; r < 8; ++r)
          orow[(size_t)(16 * mi + half * 8 + r) * NOUT + n] = memo[r];
      }
    }

    // ---- pre-barrier: prepare input drive for step t+1 ------------------
    if (t + 1 < NT) {
      if (USE_WS) {
        const float* base =
            cur + ((((size_t)(t + 1) * 8 + mi) * 4 + ni0) * 32 + lane) * 8;
        curN[0] = syn[0] * a_h + *(const v8f*)(base);
        curN[1] = syn[1] * a_h + *(const v8f*)(base + 32 * 8);
        __builtin_prefetch(base + 8 * 4 * 32 * 8, 0, 1);       // slot t+2
        __builtin_prefetch(base + 8 * 4 * 32 * 8 + 64, 0, 1);
      } else {
        const float* xrow =
            x + (size_t)t * NB * NIN + (size_t)(16 * mi + n) * NIN;
        v16bf xf[3];
#pragma unroll
        for (int c = 0; c < 3; ++c) xf[c] = afrag_from_global(xrow, 32 * c + half * 16);
        __builtin_prefetch(xrow + NB * NIN, 0, 1);
        __builtin_prefetch(xrow + NB * NIN + 32, 0, 1);
        __builtin_prefetch(xrow + NB * NIN + 64, 0, 1);
#pragma unroll
        for (int j = 0; j < 2; ++j) {
          v8f a = syn[j] * a_h;
          a = wmma_bf16(xf[0], whf[j][0], a);
          a = wmma_bf16(xf[1], whf[j][1], a);
          a = wmma_bf16(xf[2], whf[j][2], a);
          curN[j] = a;
        }
      }
    }

    __syncthreads();               // one barrier per timestep
  }
}

extern "C" void kernel_launch(void* const* d_in, const int* in_sizes, int n_in,
                              void* d_out, int out_size, void* d_ws, size_t ws_size,
                              hipStream_t stream) {
  (void)in_sizes; (void)n_in; (void)out_size;
  const float* x  = (const float*)d_in[0];
  const float* Wh = (const float*)d_in[1];
  const float* Vw = (const float*)d_in[2];
  const float* Wo = (const float*)d_in[3];
  float* out = (float*)d_out;

  const float sp1 = log1pf(expf(1.0f));            // softplus(1)
  const float a_h = expf(-0.004f / (sp1 * 0.01f)); // alpha (syn decay)
  const float b_h = expf(-0.004f / (sp1 * 0.02f)); // beta  (mem decay)
  const float a_o = a_h, b_o = b_h;

  const size_t need = (size_t)NT * NB * NH * sizeof(float);   // 128 MB
  if (ws_size >= need) {
    float* cur = (float*)d_ws;
    tinyrsnn_input_gemm<<<NT, 256, 0, stream>>>(x, Wh, cur);
    tinyrsnn_scan<true><<<1, 512, 0, stream>>>(x, Wh, Vw, Wo, cur, out,
                                               a_h, b_h, a_o, b_o);
  } else {
    tinyrsnn_scan<false><<<1, 512, 0, stream>>>(x, Wh, Vw, Wo, nullptr, out,
                                                a_h, b_h, a_o, b_o);
  }
}
